// BPPSLodeModel_18081812316536
// MI455X (gfx1250) — matrix operand order repaired and checked by reading, not compiled, not run on hardware
//
#include <hip/hip_runtime.h>

// MI455X (gfx1250) wave32 WMMA implementation, M=32 per wave.
// Roofline: 410 MB of fp32 features @ 23.3 TB/s => ~18 us floor. Species
// bucketing keeps matrix work 1x (no 4x species redundancy); each wave does a
// species-pure 32-atom tile so every f16 B-fragment load from L2 feeds two
// A-fragments (L2 weight traffic ~450 MB total, well under the HBM stream).

typedef __attribute__((ext_vector_type(16))) _Float16 v16h;
typedef __attribute__((ext_vector_type(8)))  _Float16 v8h;
typedef __attribute__((ext_vector_type(8)))  float    v8f;

#define NSPEC 4
#define FPS   256
#define LNEPS 1e-5f

// ---------------------------------------------------------------- k_init ---
__global__ void k_init(int* __restrict__ cnt, int* __restrict__ bucket, long long nb) {
  long long i = (long long)blockIdx.x * blockDim.x + threadIdx.x;
  if (i < NSPEC) cnt[i] = 0;
  for (; i < nb; i += (long long)gridDim.x * blockDim.x) bucket[i] = -1;
}

// -------------------------------------------------------------- k_bucket ---
// Ballot-based per-wave slot allocation: one atomicAdd per (wave, species).
// Bucket order is non-deterministic but per-atom energies are grouping-
// invariant (rows never mix), so the final reduced output is deterministic.
__global__ void k_bucket(const int* __restrict__ species, int* __restrict__ cnt,
                         int* __restrict__ bucket, int N, int stride) {
  int i = blockIdx.x * blockDim.x + threadIdx.x;
  int lane = threadIdx.x & 31;
  int sp = (i < N) ? species[i] : -1;
  for (int s = 0; s < NSPEC; ++s) {
    unsigned mm = (unsigned)__ballot(sp == s);   // wave32: low 32 bits
    if (!mm) continue;                           // wave-uniform
    int cw = __popc(mm);
    int leader = __ffs(mm) - 1;
    int base = 0;
    if (lane == leader) base = atomicAdd(&cnt[s], cw);
    base = __shfl(base, leader);
    if (sp == s) {
      int rank = __popc(mm & ((1u << lane) - 1u));
      bucket[s * stride + base + rank] = i;
    }
  }
}

// -------------------------------------------------------------- k_wprep ---
// Convert W_ps1 [4,256,64] and W_ps2 [4,64,32] fp32 -> f16 WMMA B-fragments.
// Fragment = 32 lanes x 16 f16 (32B per lane, lane-contiguous => b128 loads).
// B(32x16): lane<16 holds col=lane, K=kb*32+e+(e>=8?8:0); lane>=16 col=lane-16,
// K=kb*32+8+e+(e>=8?8:0)  (CDNA5 16-bit B striping).
__global__ void k_wprep(const float* __restrict__ W1, const float* __restrict__ W2,
                        v16h* __restrict__ wf) {
  int gid = blockIdx.x * blockDim.x + threadIdx.x;
  int f = gid >> 5, lane = gid & 31;
  if (f >= 144) return;
  v16h t;
  if (f < 128) {           // W_ps1: s(4) x nt(4) x kb(8)
    int s = f >> 5, rem = f & 31, nt = rem >> 3, kb = rem & 7;
    int kbase = kb * 32 + ((lane < 16) ? 0 : 8);
    int n = nt * 16 + (lane & 15);
#pragma unroll
    for (int e = 0; e < 16; ++e)
      t[e] = (_Float16)W1[(s * FPS + kbase + e + (e & 8)) * 64 + n];
  } else {                 // W_ps2: s(4) x nt(2) x kb(2)
    int f2 = f - 128;
    int s = f2 >> 2, rem = f2 & 3, nt = rem >> 1, kb = rem & 1;
    int kbase = kb * 32 + ((lane < 16) ? 0 : 8);
    int n = nt * 16 + (lane & 15);
#pragma unroll
    for (int e = 0; e < 16; ++e)
      t[e] = (_Float16)W2[(s * 64 + kbase + e + (e & 8)) * 32 + n];
  }
  wf[f * 32 + lane] = t;
}

// ---------------------------------------------------------------- k_main ---
// One wave per species-pure 32-atom tile (two 16-row M-tiles per B fragment).
__global__ void __launch_bounds__(256)
k_main(const float* __restrict__ fps, const float* __restrict__ fmp,
       const int* __restrict__ cnt, const int* __restrict__ bucket,
       const v16h* __restrict__ wf, float* __restrict__ e_atom,
       const float* __restrict__ b1, const float* __restrict__ g1, const float* __restrict__ be1,
       const float* __restrict__ b2, const float* __restrict__ g2, const float* __restrict__ be2,
       const float* __restrict__ W3, const float* __restrict__ b3,
       const float* __restrict__ Wmp1, const float* __restrict__ bmp1,
       const float* __restrict__ gmp1, const float* __restrict__ bemp1,
       const float* __restrict__ Wmp2, const float* __restrict__ bmp2,
       const float* __restrict__ gmp2, const float* __restrict__ bemp2,
       const float* __restrict__ Wmp3, const float* __restrict__ bmp3,
       const float* __restrict__ compw, int N, int stride) {
  __shared__ float P[3632];
  __shared__ __align__(16) _Float16 HB[8][2 * 16 * 72];  // per-wave, per-mtile

  auto cp = [&](const float* src, int dst, int len) {
    for (int i = threadIdx.x; i < len; i += blockDim.x) P[dst + i] = src[i];
  };
  cp(b1, 0, 256);     cp(g1, 256, 64);    cp(be1, 320, 64);
  cp(b2, 384, 128);   cp(g2, 512, 32);    cp(be2, 544, 32);
  cp(W3, 576, 128);   cp(b3, 704, 4);
  cp(Wmp1, 708, 512); cp(bmp1, 1220, 128); cp(gmp1, 1348, 32); cp(bemp1, 1380, 32);
  cp(Wmp2, 1412, 2048); cp(bmp2, 3460, 64); cp(gmp2, 3524, 16); cp(bemp2, 3540, 16);
  cp(Wmp3, 3556, 64); cp(bmp3, 3620, 4);  cp(compw, 3624, 4);
  __syncthreads();

  int lane = threadIdx.x & 31;
  int wslot = threadIdx.x >> 5;
  int tile = blockIdx.x * 8 + wslot;

  int c0 = cnt[0], c1 = cnt[1], c2 = cnt[2], c3 = cnt[3];
  int t0 = (c0 + 31) >> 5, t1 = (c1 + 31) >> 5, t2 = (c2 + 31) >> 5, t3 = (c3 + 31) >> 5;
  int s, ts;
  if      (tile < t0)                { s = 0; ts = tile; }
  else if (tile < t0 + t1)           { s = 1; ts = tile - t0; }
  else if (tile < t0 + t1 + t2)      { s = 2; ts = tile - t0 - t1; }
  else if (tile < t0 + t1 + t2 + t3) { s = 3; ts = tile - t0 - t1 - t2; }
  else return;                        // past the only barrier: safe exit

  const int* bidx = bucket + s * stride + ts * 32;
  int col = lane & 15;
  // Padding rows (bucket entry -1) just recompute row 0's data; rows never
  // mix (per-row LN via half-wave shfl), so validity only gates the store.
  int aid0 = bidx[col], aid1 = bidx[16 + col];
  const float4* ar0 = (const float4*)(fps + (long long)(aid0 >= 0 ? aid0 : 0) * FPS);
  const float4* ar1 = (const float4*)(fps + (long long)(aid1 >= 0 ? aid1 : 0) * FPS);

  // ---------------- layer 1: [32x256] x [256x64] via 8x4x2 WMMA -----------
  v8f z = {0.f, 0.f, 0.f, 0.f, 0.f, 0.f, 0.f, 0.f};
  v8f acc[2][4] = {{z, z, z, z}, {z, z, z, z}};
#pragma unroll
  for (int kb = 0; kb < 8; ++kb) {
    int k0 = kb * 32 + ((lane < 16) ? 0 : 8);     // A 16-bit layout (16x32)
    float4 fa = ar0[(k0 >> 2) + 0], fb = ar0[(k0 >> 2) + 1];
    float4 fc = ar0[(k0 >> 2) + 4], fd = ar0[(k0 >> 2) + 5];
    float4 ga = ar1[(k0 >> 2) + 0], gb = ar1[(k0 >> 2) + 1];
    float4 gc = ar1[(k0 >> 2) + 4], gd = ar1[(k0 >> 2) + 5];
    v16h a0, a1;
    a0[0] = (_Float16)fa.x; a0[1] = (_Float16)fa.y; a0[2] = (_Float16)fa.z; a0[3] = (_Float16)fa.w;
    a0[4] = (_Float16)fb.x; a0[5] = (_Float16)fb.y; a0[6] = (_Float16)fb.z; a0[7] = (_Float16)fb.w;
    a0[8]  = (_Float16)fc.x; a0[9]  = (_Float16)fc.y; a0[10] = (_Float16)fc.z; a0[11] = (_Float16)fc.w;
    a0[12] = (_Float16)fd.x; a0[13] = (_Float16)fd.y; a0[14] = (_Float16)fd.z; a0[15] = (_Float16)fd.w;
    a1[0] = (_Float16)ga.x; a1[1] = (_Float16)ga.y; a1[2] = (_Float16)ga.z; a1[3] = (_Float16)ga.w;
    a1[4] = (_Float16)gb.x; a1[5] = (_Float16)gb.y; a1[6] = (_Float16)gb.z; a1[7] = (_Float16)gb.w;
    a1[8]  = (_Float16)gc.x; a1[9]  = (_Float16)gc.y; a1[10] = (_Float16)gc.z; a1[11] = (_Float16)gc.w;
    a1[12] = (_Float16)gd.x; a1[13] = (_Float16)gd.y; a1[14] = (_Float16)gd.z; a1[15] = (_Float16)gd.w;
#pragma unroll
    for (int nt = 0; nt < 4; ++nt) {
      v16h bf = wf[((s * 4 + nt) * 8 + kb) * 32 + lane];   // reused by both M-tiles
      acc[0][nt] = __builtin_amdgcn_wmma_f32_16x16x32_f16(false, a0, false, bf,
                                                          (short)0, acc[0][nt], false, false);
      acc[1][nt] = __builtin_amdgcn_wmma_f32_16x16x32_f16(false, a1, false, bf,
                                                          (short)0, acc[1][nt], false, false);
    }
  }

  // bias + LayerNorm(64) + ReLU; stats via half-wave shfl on C-fragments
  float gv[4], bev[4], biv[4];
#pragma unroll
  for (int nt = 0; nt < 4; ++nt) {
    int c = nt * 16 + col;
    gv[nt] = P[256 + c]; bev[nt] = P[320 + c]; biv[nt] = P[s * 64 + c];
  }
#pragma unroll
  for (int mt = 0; mt < 2; ++mt) {
#pragma unroll
    for (int r = 0; r < 8; ++r) {
      float x[4], sm = 0.f, sq = 0.f;
#pragma unroll
      for (int nt = 0; nt < 4; ++nt) {
        x[nt] = acc[mt][nt][r] + biv[nt]; sm += x[nt]; sq += x[nt] * x[nt];
      }
#pragma unroll
      for (int o = 1; o < 16; o <<= 1) { sm += __shfl_xor(sm, o); sq += __shfl_xor(sq, o); }
      float mean = sm * (1.f / 64.f);
      float rs = rsqrtf(sq * (1.f / 64.f) - mean * mean + LNEPS);
      int m = (lane < 16) ? r : r + 8;
#pragma unroll
      for (int nt = 0; nt < 4; ++nt) {
        float h = fmaxf((x[nt] - mean) * rs * gv[nt] + bev[nt], 0.f);
        HB[wslot][mt * 1152 + m * 72 + nt * 16 + col] = (_Float16)h;  // 72 f16/row pad
      }
    }
  }
  __builtin_amdgcn_wave_barrier();   // keep DS store->load order within wave

  // ---------------- layer 2: [32x64] x [64x32] via 2x2x2 WMMA -------------
  v8f acc2[2][2] = {{z, z}, {z, z}};
#pragma unroll
  for (int kb = 0; kb < 2; ++kb) {
    int k0 = kb * 32 + ((lane < 16) ? 0 : 8);
    v16h a2[2];
#pragma unroll
    for (int mt = 0; mt < 2; ++mt) {
      const _Float16* hrow = &HB[wslot][mt * 1152 + col * 72];
      v8h lo = *(const v8h*)(hrow + k0);
      v8h hi = *(const v8h*)(hrow + k0 + 16);
#pragma unroll
      for (int e = 0; e < 8; ++e) { a2[mt][e] = lo[e]; a2[mt][8 + e] = hi[e]; }
    }
#pragma unroll
    for (int nt = 0; nt < 2; ++nt) {
      v16h bf = wf[(128 + s * 4 + nt * 2 + kb) * 32 + lane];
      acc2[0][nt] = __builtin_amdgcn_wmma_f32_16x16x32_f16(false, a2[0], false, bf,
                                                           (short)0, acc2[0][nt], false, false);
      acc2[1][nt] = __builtin_amdgcn_wmma_f32_16x16x32_f16(false, a2[1], false, bf,
                                                           (short)0, acc2[1][nt], false, false);
    }
  }

  // bias + LayerNorm(32) + ReLU + layer-3 dot (32 -> 1) on fragments
  float b2v0 = P[384 + s * 32 + col], b2v1 = P[384 + s * 32 + 16 + col];
  float g2v0 = P[512 + col], g2v1 = P[512 + 16 + col];
  float be2v0 = P[544 + col], be2v1 = P[544 + 16 + col];
  float w3v0 = P[576 + s * 32 + col], w3v1 = P[576 + s * 32 + 16 + col];
  float esel[2] = {0.f, 0.f};
#pragma unroll
  for (int mt = 0; mt < 2; ++mt) {
#pragma unroll
    for (int r = 0; r < 8; ++r) {
      float x0 = acc2[mt][0][r] + b2v0, x1 = acc2[mt][1][r] + b2v1;
      float sm = x0 + x1, sq = x0 * x0 + x1 * x1;
#pragma unroll
      for (int o = 1; o < 16; o <<= 1) { sm += __shfl_xor(sm, o); sq += __shfl_xor(sq, o); }
      float mean = sm * (1.f / 32.f);
      float rs = rsqrtf(sq * (1.f / 32.f) - mean * mean + LNEPS);
      float h0 = fmaxf((x0 - mean) * rs * g2v0 + be2v0, 0.f);
      float h1 = fmaxf((x1 - mean) * rs * g2v1 + be2v1, 0.f);
      float d = h0 * w3v0 + h1 * w3v1;
#pragma unroll
      for (int o = 1; o < 16; o <<= 1) d += __shfl_xor(d, o);
      if (col == r) esel[mt] = d;   // lanes 0-7: rows 0-7; lanes 16-23: rows 8-15
    }
  }

  // ---------------- mesh-potential branch + final store (holder lanes) ----
  bool holder = col < 8;
  int row = col + ((lane < 16) ? 0 : 8);
  auto mpE = [&](int atom) -> float {
    float x0f = fmp[atom * 4 + 0], x1f = fmp[atom * 4 + 1];
    float x2f = fmp[atom * 4 + 2], x3f = fmp[atom * 4 + 3];
    float h1m[32]; float sm = 0.f, sq = 0.f;
#pragma unroll
    for (int o = 0; o < 32; ++o) {
      float t = P[1220 + s * 32 + o]
              + x0f * P[708 + (s * 4 + 0) * 32 + o]
              + x1f * P[708 + (s * 4 + 1) * 32 + o]
              + x2f * P[708 + (s * 4 + 2) * 32 + o]
              + x3f * P[708 + (s * 4 + 3) * 32 + o];
      h1m[o] = t; sm += t; sq += t * t;
    }
    float mean = sm * (1.f / 32.f);
    float rs = rsqrtf(sq * (1.f / 32.f) - mean * mean + LNEPS);
#pragma unroll
    for (int o = 0; o < 32; ++o)
      h1m[o] = fmaxf((h1m[o] - mean) * rs * P[1348 + o] + P[1380 + o], 0.f);
    float h2m[16]; sm = 0.f; sq = 0.f;
#pragma unroll
    for (int j = 0; j < 16; ++j) {
      float t = P[3460 + s * 16 + j];
#pragma unroll
      for (int o = 0; o < 32; ++o) t += h1m[o] * P[1412 + (s * 32 + o) * 16 + j];
      h2m[j] = t; sm += t; sq += t * t;
    }
    mean = sm * (1.f / 16.f);
    rs = rsqrtf(sq * (1.f / 16.f) - mean * mean + LNEPS);
    float emp = P[3620 + s];
#pragma unroll
    for (int j = 0; j < 16; ++j)
      emp += fmaxf((h2m[j] - mean) * rs * P[3524 + j] + P[3540 + j], 0.f) * P[3556 + s * 16 + j];
    return emp;
  };
#pragma unroll
  for (int mt = 0; mt < 2; ++mt) {
    int aidh = holder ? bidx[mt * 16 + row] : -1;
    if (holder && aidh >= 0) {
      float e = esel[mt] + P[704 + s] + P[3624 + s];  // b_ps3 + comp_w term
      e_atom[aidh] = e + mpE(aidh);
    }
  }
}

// -------------------------------------------------------------- k_reduce ---
// Deterministic per-structure segment sum (batch is sorted).
__global__ void k_reduce(const int* __restrict__ batch, const float* __restrict__ e_atom,
                         float* __restrict__ out, int N) {
  int b = blockIdx.x;
  auto lb = [&](int key) {
    int lo = 0, hi = N;
    while (lo < hi) { int mid = (lo + hi) >> 1; if (batch[mid] < key) lo = mid + 1; else hi = mid; }
    return lo;
  };
  int lo = lb(b), hi = lb(b + 1);
  float sum = 0.f;
  for (int i = lo + threadIdx.x; i < hi; i += blockDim.x) sum += e_atom[i];
#pragma unroll
  for (int o = 16; o >= 1; o >>= 1) sum += __shfl_xor(sum, o);
  __shared__ float red[8];
  int lane = threadIdx.x & 31, w = threadIdx.x >> 5;
  if (lane == 0) red[w] = sum;
  __syncthreads();
  if (threadIdx.x < 8) {
    float v = red[threadIdx.x];
#pragma unroll
    for (int o = 4; o >= 1; o >>= 1) v += __shfl_xor(v, o);
    if (threadIdx.x == 0) out[b] = v;
  }
}

// ---------------------------------------------------------------- launch ---
extern "C" void kernel_launch(void* const* d_in, const int* in_sizes, int n_in,
                              void* d_out, int out_size, void* d_ws, size_t ws_size,
                              hipStream_t stream) {
  const float* fps  = (const float*)d_in[0];
  const float* fmp  = (const float*)d_in[1];
  const int*   spec = (const int*)d_in[2];
  const int*   bat  = (const int*)d_in[3];
  const float* W1   = (const float*)d_in[4];
  const float* b1   = (const float*)d_in[5];
  const float* g1   = (const float*)d_in[6];
  const float* be1  = (const float*)d_in[7];
  const float* W2   = (const float*)d_in[8];
  const float* b2   = (const float*)d_in[9];
  const float* g2   = (const float*)d_in[10];
  const float* be2  = (const float*)d_in[11];
  const float* W3   = (const float*)d_in[12];
  const float* b3   = (const float*)d_in[13];
  const float* Wmp1 = (const float*)d_in[14];
  const float* bmp1 = (const float*)d_in[15];
  const float* gmp1 = (const float*)d_in[16];
  const float* bemp1= (const float*)d_in[17];
  const float* Wmp2 = (const float*)d_in[18];
  const float* bmp2 = (const float*)d_in[19];
  const float* gmp2 = (const float*)d_in[20];
  const float* bemp2= (const float*)d_in[21];
  const float* Wmp3 = (const float*)d_in[22];
  const float* bmp3 = (const float*)d_in[23];
  const float* cw   = (const float*)d_in[24];

  int N = in_sizes[2];
  int B = out_size;
  int stride = (N + 31) & ~31;            // per-species bucket stride, 32-padded

  char* ws = (char*)d_ws;
  int*  cnt    = (int*)ws;                                   // 4 ints
  int*  bucket = (int*)(ws + 256);                           // 4*stride ints
  size_t off_e = 256 + (((size_t)4 * stride * sizeof(int) + 255) & ~(size_t)255);
  float* e_atom = (float*)(ws + off_e);                      // N floats
  size_t off_wf = off_e + (((size_t)N * sizeof(float) + 255) & ~(size_t)255);
  v16h* wfrag = (v16h*)(ws + off_wf);                        // 144 frags * 1 KiB

  long long nb = (long long)4 * stride;
  k_init<<<2048, 256, 0, stream>>>(cnt, bucket, nb);
  k_bucket<<<(N + 255) / 256, 256, 0, stream>>>(spec, cnt, bucket, N, stride);
  k_wprep<<<18, 256, 0, stream>>>(W1, W2, wfrag);

  int maxtiles = (N + 31) / 32 + NSPEC;   // worst-case per-species padding
  int blocks = (maxtiles + 7) / 8;        // 8 waves (tiles) per 256-thread block
  k_main<<<blocks, 256, 0, stream>>>(fps, fmp, cnt, bucket, wfrag, e_atom,
                                     b1, g1, be1, b2, g2, be2, W3, b3,
                                     Wmp1, bmp1, gmp1, bemp1,
                                     Wmp2, bmp2, gmp2, bemp2,
                                     Wmp3, bmp3, cw, N, stride);
  k_reduce<<<B, 256, 0, stream>>>(bat, e_atom, (float*)d_out, N);
}
